// GLANT_68865505624454
// MI455X (gfx1250) — matrix operand order repaired and checked by reading, not compiled, not run on hardware
//
#include <hip/hip_runtime.h>
#include <cstdint>
#include <cmath>

// Problem constants (match reference)
#define N_NODES 50000
#define EDGES   400000
#define DIN     128
#define DHID    128
#define DOUT    64
#define HTC     4      // TransConv heads

typedef __attribute__((ext_vector_type(16))) _Float16 v16h;
typedef __attribute__((ext_vector_type(8)))  float    v8f;

// ---------------------------------------------------------------- utilities

__global__ void glant_fill_f32(float* __restrict__ p, float v, int n) {
  int i = blockIdx.x * blockDim.x + threadIdx.x;
  if (i < n) p[i] = v;
}

__global__ void glant_cast_f16(const float* __restrict__ src, _Float16* __restrict__ dst, int n) {
  int i = blockIdx.x * blockDim.x + threadIdx.x;
  if (i < n) dst[i] = (_Float16)src[i];
}

__global__ void glant_elu_cast(const float* __restrict__ src, _Float16* __restrict__ dst, int n) {
  int i = blockIdx.x * blockDim.x + threadIdx.x;
  if (i < n) {
    float x = src[i];
    float y = x > 0.0f ? x : expm1f(x);
    dst[i] = (_Float16)y;
  }
}

// Pack Wq|Wk|Wv ([K,D] f32 each) into a transposed f16 panel Wt[3D, K]
// (column-contiguous for the WMMA B operand), and pack the three biases.
__global__ void glant_pack_qkv(const float* __restrict__ Wq, const float* __restrict__ Wk,
                               const float* __restrict__ Wv,
                               const float* __restrict__ bq, const float* __restrict__ bk,
                               const float* __restrict__ bv,
                               _Float16* __restrict__ w16t, float* __restrict__ bpack,
                               int K, int D) {
  int threeD = 3 * D;
  int total  = K * threeD;
  int idx = blockIdx.x * blockDim.x + threadIdx.x;
  if (idx < total) {
    int k = idx / threeD;
    int n = idx % threeD;
    const float* src; int nn;
    if (n < D)            { src = Wq; nn = n; }
    else if (n < 2 * D)   { src = Wk; nn = n - D; }
    else                  { src = Wv; nn = n - 2 * D; }
    w16t[(size_t)n * K + k] = (_Float16)src[(size_t)k * D + nn];
  }
  if (idx < threeD) {
    int n = idx;
    const float* sb; int nn;
    if (n < D)          { sb = bq; nn = n; }
    else if (n < 2 * D) { sb = bk; nn = n - D; }
    else                { sb = bv; nn = n - 2 * D; }
    bpack[n] = sb[nn];
  }
}

// ---------------------------------------------------------------- WMMA GEMM
// C[nrows, ncols] = A[nrows, 128](f16, row-major) * Wt^T + bias
// Wt stored as [ncols, 128] (each output column's K-vector contiguous).
// One wave32 computes one 16x16 output tile; K loop = 4 x v_wmma_f32_16x16x32_f16.
__global__ void __launch_bounds__(256)
glant_wmma_gemm(const _Float16* __restrict__ A, const _Float16* __restrict__ Wt,
                const float* __restrict__ bias, float* __restrict__ C,
                int nrows, int ncols) {
  constexpr int K = 128;
  const int wave = threadIdx.x >> 5;
  const int lane = threadIdx.x & 31;
  const int tilesN = ncols >> 4;
  const int totalTiles = (nrows >> 4) * tilesN;
  const int tile = blockIdx.x * 8 + wave;   // wave-uniform -> EXEC stays all-ones for WMMA
  if (tile >= totalTiles) return;

  const int m0 = (tile / tilesN) << 4;
  const int n0 = (tile % tilesN) << 4;
  const int hi = lane >> 4;     // half-wave select (ISA A/B layout)
  const int lr = lane & 15;

  const _Float16* __restrict__ arow = A  + (size_t)(m0 + lr) * K;  // A: M = lane%16
  const _Float16* __restrict__ bcol = Wt + (size_t)(n0 + lr) * K;  // B: N = lane%16

  v8f acc = {};
#pragma unroll
  for (int k0 = 0; k0 < K; k0 += 32) {
    v16h a, b;
    // A 16x32 f16 layout: lanes 0-15 -> K = k0+0..7 | k0+16..23 ; lanes 16-31 -> +8
#pragma unroll
    for (int i = 0; i < 8; ++i) {
      a[i]     = arow[k0 + hi * 8 + i];
      a[i + 8] = arow[k0 + 16 + hi * 8 + i];
    }
    // B 32x16 f16 layout: lanes 0-15 -> K = k0+0..7 | k0+16..23 ; lanes 16-31 -> +8
#pragma unroll
    for (int i = 0; i < 8; ++i) {
      b[i]     = bcol[k0 + hi * 8 + i];
      b[i + 8] = bcol[k0 + 16 + hi * 8 + i];
    }
    acc = __builtin_amdgcn_wmma_f32_16x16x32_f16(
        /*neg_a=*/false, a, /*neg_b=*/false, b,
        /*c_mod=*/(short)0, acc, /*reuse_a=*/false, /*reuse_b=*/false);
  }

  const int colg = n0 + lr;
  const float bb = bias[colg];
  // D layout: VGPR r -> row m0 + r + hi*8, col n0 + lane%16
#pragma unroll
  for (int r = 0; r < 8; ++r) {
    int rowg = m0 + r + hi * 8;
    C[(size_t)rowg * ncols + colg] = acc[r] + bb;
  }
}

// ---------------------------------------------------------------- edge phases

__device__ inline void glant_atomic_max_f32(float* addr, float val) {
  int* ia = (int*)addr;
  int old = __float_as_int(*addr);
  while (__int_as_float(old) < val) {
    int assumed = old;
    old = atomicCAS(ia, assumed, __float_as_int(val));
    if (old == assumed) break;
  }
}

// s[e,h] = <q[row[e],h,:], k[col[e],h,:]> * scale ; segment-max into mbuf[row,h]
__global__ void glant_edge_score(const float* __restrict__ qkv,
                                 const int* __restrict__ rowp, const int* __restrict__ colp,
                                 float* __restrict__ sbuf, float* __restrict__ mbuf,
                                 int nE, int D, int dk, float scale) {
  int idx = blockIdx.x * blockDim.x + threadIdx.x;
  if (idx >= nE * HTC) return;
  int e = idx >> 2, h = idx & 3;
  int r = rowp[e], c = colp[e];
  const float* __restrict__ q  = qkv + (size_t)r * 3 * D + h * dk;
  const float* __restrict__ kk = qkv + (size_t)c * 3 * D + D + h * dk;
  float acc = 0.0f;
  for (int t = 0; t < dk; ++t) acc = fmaf(q[t], kk[t], acc);
  acc *= scale;
  sbuf[idx] = acc;
  glant_atomic_max_f32(&mbuf[r * HTC + h], acc);
}

// e[e,h] = exp(s - m[row]); segment-sum into dbuf[row,h]
__global__ void glant_edge_exp(const int* __restrict__ rowp,
                               float* __restrict__ sbuf, const float* __restrict__ mbuf,
                               float* __restrict__ dbuf, int nE) {
  int idx = blockIdx.x * blockDim.x + threadIdx.x;
  if (idx >= nE * HTC) return;
  int e = idx >> 2, h = idx & 3;
  int r = rowp[e];
  float ev = expf(sbuf[idx] - mbuf[r * HTC + h]);
  sbuf[idx] = ev;
  atomicAdd(&dbuf[r * HTC + h], ev);
}

// acc[row, f] += beta * (e/denom) * v[col, f]
__global__ void glant_edge_agg(const float* __restrict__ qkv,
                               const int* __restrict__ rowp, const int* __restrict__ colp,
                               const float* __restrict__ sbuf, const float* __restrict__ dbuf,
                               float* __restrict__ acc, int nE, int D, int dk, float beta) {
  long long idx = (long long)blockIdx.x * blockDim.x + threadIdx.x;
  long long total = (long long)nE * D;
  if (idx >= total) return;
  int e = (int)(idx / D);
  int f = (int)(idx % D);
  int h = f / dk;
  int r = rowp[e], c = colp[e];
  float attn = sbuf[e * HTC + h] / (dbuf[r * HTC + h] + 1e-16f);
  float vv = qkv[(size_t)c * 3 * D + 2 * D + f];
  atomicAdd(&acc[(size_t)r * D + f], beta * attn * vv);
}

// ---------------------------------------------------------------- log-softmax (64 cols, wave32)

__global__ void __launch_bounds__(256)
glant_log_softmax64(const float* __restrict__ in, float* __restrict__ out, int n) {
  int wave = threadIdx.x >> 5;
  int lane = threadIdx.x & 31;
  int node = blockIdx.x * 8 + wave;
  if (node >= n) return;
  const float* x = in + (size_t)node * 64;
  float a = x[lane];
  float b = x[lane + 32];
  float mx = fmaxf(a, b);
#pragma unroll
  for (int off = 16; off > 0; off >>= 1) mx = fmaxf(mx, __shfl_xor(mx, off, 32));
  float s = expf(a - mx) + expf(b - mx);
#pragma unroll
  for (int off = 16; off > 0; off >>= 1) s += __shfl_xor(s, off, 32);
  float lse = mx + logf(s);
  out[(size_t)node * 64 + lane]      = a - lse;
  out[(size_t)node * 64 + lane + 32] = b - lse;
}

// ---------------------------------------------------------------- driver

extern "C" void kernel_launch(void* const* d_in, const int* in_sizes, int n_in,
                              void* d_out, int out_size, void* d_ws, size_t ws_size,
                              hipStream_t stream) {
  (void)in_sizes; (void)n_in; (void)out_size; (void)ws_size;

  const float* x   = (const float*)d_in[0];
  const int* edge0 = (const int*)d_in[1];
  const int* edge1 = (const int*)d_in[2];
  const float* W0q = (const float*)d_in[3];
  const float* W0k = (const float*)d_in[4];
  const float* W0v = (const float*)d_in[5];
  const float* b0q = (const float*)d_in[6];
  const float* b0k = (const float*)d_in[7];
  const float* b0v = (const float*)d_in[8];
  const float* W1q = (const float*)d_in[9];
  const float* W1k = (const float*)d_in[10];
  const float* W1v = (const float*)d_in[11];
  const float* b1q = (const float*)d_in[12];
  const float* b1k = (const float*)d_in[13];
  const float* b1v = (const float*)d_in[14];
  float* out = (float*)d_out;

  // workspace carve-out (256B aligned)
  char* base = (char*)d_ws;
  size_t off = 0;
  auto alloc = [&](size_t bytes) -> void* {
    off = (off + 255) & ~(size_t)255;
    void* p = base + off;
    off += bytes;
    return p;
  };
  _Float16* x16  = (_Float16*)alloc((size_t)N_NODES * DIN * 2);
  _Float16* h16  = (_Float16*)alloc((size_t)N_NODES * DHID * 2);
  _Float16* w16t = (_Float16*)alloc((size_t)3 * DHID * DIN * 2);
  float* bpack = (float*)alloc((size_t)3 * DHID * 4);
  float* qkv   = (float*)alloc((size_t)N_NODES * 3 * DHID * 4);
  float* sbuf  = (float*)alloc((size_t)EDGES * HTC * 4);
  float* mbuf  = (float*)alloc((size_t)N_NODES * HTC * 4);
  float* dbuf  = (float*)alloc((size_t)N_NODES * HTC * 4);
  float* acc0  = (float*)alloc((size_t)N_NODES * DHID * 4);
  float* acc1  = (float*)alloc((size_t)N_NODES * DOUT * 4);

  const int TB = 256;
  auto blocks = [](long long n, int tb) { return (unsigned)((n + tb - 1) / tb); };

  glant_cast_f16<<<blocks((long long)N_NODES * DIN, TB), TB, 0, stream>>>(x, x16, N_NODES * DIN);
  glant_fill_f32<<<blocks((long long)N_NODES * DHID, TB), TB, 0, stream>>>(acc0, 0.0f, N_NODES * DHID);

  // ---------------- layer 0: NH0=2 heads x KHOPS=2 hops, D=128, dk=32
  {
    const int D = DHID, dk = DHID / HTC;
    const float scale = 1.0f / sqrtf((float)dk);
    const int tiles = (N_NODES / 16) * ((3 * D) / 16);
    for (int hd = 0; hd < 2; ++hd) {
      for (int hop = 0; hop < 2; ++hop) {
        int s = hd * 2 + hop;
        const float* wq = W0q + (size_t)s * DIN * D;
        const float* wk = W0k + (size_t)s * DIN * D;
        const float* wv = W0v + (size_t)s * DIN * D;
        const float* bq = b0q + (size_t)s * D;
        const float* bk = b0k + (size_t)s * D;
        const float* bv = b0v + (size_t)s * D;
        glant_pack_qkv<<<blocks((long long)DIN * 3 * D, TB), TB, 0, stream>>>(
            wq, wk, wv, bq, bk, bv, w16t, bpack, DIN, D);
        glant_wmma_gemm<<<(tiles + 7) / 8, 256, 0, stream>>>(x16, w16t, bpack, qkv, N_NODES, 3 * D);

        glant_fill_f32<<<blocks((long long)N_NODES * HTC, TB), TB, 0, stream>>>(mbuf, -3.0e38f, N_NODES * HTC);
        glant_fill_f32<<<blocks((long long)N_NODES * HTC, TB), TB, 0, stream>>>(dbuf, 0.0f, N_NODES * HTC);

        const int* edges = (hop == 0) ? edge0 : edge1;
        const int* rowp = edges;
        const int* colp = edges + EDGES;
        glant_edge_score<<<blocks((long long)EDGES * HTC, TB), TB, 0, stream>>>(
            qkv, rowp, colp, sbuf, mbuf, EDGES, D, dk, scale);
        glant_edge_exp<<<blocks((long long)EDGES * HTC, TB), TB, 0, stream>>>(
            rowp, sbuf, mbuf, dbuf, EDGES);
        float beta = (hop == 0) ? 1.0f : 0.25f;   // BETA_MUL/(hop+1) = 0.5/2
        glant_edge_agg<<<blocks((long long)EDGES * D, TB), TB, 0, stream>>>(
            qkv, rowp, colp, sbuf, dbuf, acc0, EDGES, D, dk, beta);
      }
    }
  }

  glant_elu_cast<<<blocks((long long)N_NODES * DHID, TB), TB, 0, stream>>>(acc0, h16, N_NODES * DHID);
  glant_fill_f32<<<blocks((long long)N_NODES * DOUT, TB), TB, 0, stream>>>(acc1, 0.0f, N_NODES * DOUT);

  // ---------------- layer 1: NH1=1 head x 2 hops, D=64, dk=16
  {
    const int D = DOUT, dk = DOUT / HTC;
    const float scale = 1.0f / sqrtf((float)dk);
    const int tiles = (N_NODES / 16) * ((3 * D) / 16);
    for (int hop = 0; hop < 2; ++hop) {
      int s = hop;  // hd = 0
      const float* wq = W1q + (size_t)s * DHID * D;
      const float* wk = W1k + (size_t)s * DHID * D;
      const float* wv = W1v + (size_t)s * DHID * D;
      const float* bq = b1q + (size_t)s * D;
      const float* bk = b1k + (size_t)s * D;
      const float* bv = b1v + (size_t)s * D;
      glant_pack_qkv<<<blocks((long long)DHID * 3 * D, TB), TB, 0, stream>>>(
          wq, wk, wv, bq, bk, bv, w16t, bpack, DHID, D);
      glant_wmma_gemm<<<(tiles + 7) / 8, 256, 0, stream>>>(h16, w16t, bpack, qkv, N_NODES, 3 * D);

      glant_fill_f32<<<blocks((long long)N_NODES * HTC, TB), TB, 0, stream>>>(mbuf, -3.0e38f, N_NODES * HTC);
      glant_fill_f32<<<blocks((long long)N_NODES * HTC, TB), TB, 0, stream>>>(dbuf, 0.0f, N_NODES * HTC);

      const int* edges = (hop == 0) ? edge0 : edge1;
      const int* rowp = edges;
      const int* colp = edges + EDGES;
      glant_edge_score<<<blocks((long long)EDGES * HTC, TB), TB, 0, stream>>>(
          qkv, rowp, colp, sbuf, mbuf, EDGES, D, dk, scale);
      glant_edge_exp<<<blocks((long long)EDGES * HTC, TB), TB, 0, stream>>>(
          rowp, sbuf, mbuf, dbuf, EDGES);
      float beta = (hop == 0) ? 1.0f : 0.25f;
      glant_edge_agg<<<blocks((long long)EDGES * D, TB), TB, 0, stream>>>(
          qkv, rowp, colp, sbuf, dbuf, acc1, EDGES, D, dk, beta);
    }
  }

  glant_log_softmax64<<<(N_NODES + 7) / 8, 256, 0, stream>>>(acc1, out, N_NODES);
}
